// ResGCN3Layer_83124797046810
// MI455X (gfx1250) — compile-verified, matrix operand order checked by reading, not run on hardware
//
#include <hip/hip_runtime.h>
#include <hip/hip_bf16.h>

// ---------------------------------------------------------------------------
// ResGCN-3 on MI455X (gfx1250, wave32).
//   - Dense GEMMs: v_wmma_f32_16x16x32_bf16, one wave per 16-row block.
//   - Graph aggregation: bf16 gather + f32 global atomics (L2-resident).
//   - Elementwise fusion for D^{-1/2} norm, self-loop, bias, skip, ReLU.
// ---------------------------------------------------------------------------

typedef __attribute__((ext_vector_type(16))) __bf16 v16bf;
typedef __attribute__((ext_vector_type(8)))  __bf16 v8bf;
typedef __attribute__((ext_vector_type(8)))  float  v8f;

__device__ __forceinline__ float bf2f(unsigned short h) {
  return __uint_as_float(((unsigned)h) << 16);
}
__device__ __forceinline__ unsigned short f2bf(float f) {
  unsigned u = __float_as_uint(f);
  u += 0x7fffu + ((u >> 16) & 1u);          // round-to-nearest-even
  return (unsigned short)(u >> 16);
}

// ---------------------------- degree / dinv --------------------------------
__global__ void deg_init(float* __restrict__ deg, int n) {
  int g = blockIdx.x * blockDim.x + threadIdx.x;
  if (g < n) deg[g] = 1.0f;                  // self-loop
}
__global__ void deg_accum(const long long* __restrict__ dst, float* __restrict__ deg,
                          long long E) {
  long long g = (long long)blockIdx.x * blockDim.x + threadIdx.x;
  if (g < E) atomicAdd(&deg[(int)dst[g]], 1.0f);
}
__global__ void deg_to_dinv(float* __restrict__ deg, int n) {
  int g = blockIdx.x * blockDim.x + threadIdx.x;
  if (g < n) deg[g] = rsqrtf(deg[g]);
}

// ---------------------------- conversions ----------------------------------
__global__ void cvt_f32_to_bf16(const float* __restrict__ src,
                                unsigned short* __restrict__ dst, long long n) {
  long long g = (long long)blockIdx.x * blockDim.x + threadIdx.x;
  long long i = g * 2;
  if (i + 1 < n) {
    float2 v = *(const float2*)(src + i);
    *(unsigned*)(dst + i) = (unsigned)f2bf(v.x) | ((unsigned)f2bf(v.y) << 16);
  } else if (i < n) {
    dst[i] = f2bf(src[i]);
  }
}
// W[in][out] f32 row-major  ->  Wt[out][in] bf16 (column access for WMMA B)
__global__ void transpose_w(const float* __restrict__ W, unsigned short* __restrict__ Wt,
                            int In, int Out) {
  int g = blockIdx.x * blockDim.x + threadIdx.x;
  if (g >= In * Out) return;
  int o = g / In;
  int i = g - o * In;
  Wt[g] = f2bf(W[(size_t)i * Out + o]);
}

__global__ void zero_f32(float* __restrict__ p, long long n) {
  long long g = (long long)blockIdx.x * blockDim.x + threadIdx.x;
  if (g < n) p[g] = 0.0f;
}

// ---------------------------- WMMA GEMM ------------------------------------
// D[nrows x FOUT] (bf16) = A[nrows x FIN] (bf16) @ Wt^T  (Wt is [FOUT][FIN] bf16)
template <int FIN, int FOUT>
__global__ __launch_bounds__(128) void gemm_bf16_wmma(
    const unsigned short* __restrict__ Au, const unsigned short* __restrict__ Wtu,
    unsigned short* __restrict__ D, int nrows) {
  constexpr int KT = FIN / 32;
  constexpr int NT = FOUT / 16;
  const int wave = threadIdx.x >> 5;
  const int lane = threadIdx.x & 31;
  const int rb = blockIdx.x * 4 + wave;
  if (rb * 16 + 15 >= nrows) return;         // wave-uniform: EXEC stays all-1s
  const int m    = lane & 15;
  const int half = lane >> 4;

  // A fragment: lane<16 -> row m, K {0..7,16..23}; lane>=16 -> K {8..15,24..31}
  const unsigned short* arow = Au + (size_t)(rb * 16 + m) * FIN + half * 8;
  v16bf afrag[KT];
#pragma unroll
  for (int k = 0; k < KT; ++k) {
    v8bf lo = *(const v8bf*)(arow + k * 32);
    v8bf hi = *(const v8bf*)(arow + k * 32 + 16);
    afrag[k] = __builtin_shufflevector(lo, hi, 0, 1, 2, 3, 4, 5, 6, 7,
                                       8, 9, 10, 11, 12, 13, 14, 15);
  }

#pragma unroll
  for (int nt = 0; nt < NT; ++nt) {
    v8f acc = {};
    const unsigned short* bcol = Wtu + (size_t)(nt * 16 + m) * FIN + half * 8;
#pragma unroll
    for (int k = 0; k < KT; ++k) {
      v8bf lo = *(const v8bf*)(bcol + k * 32);
      v8bf hi = *(const v8bf*)(bcol + k * 32 + 16);
      v16bf bfrag = __builtin_shufflevector(lo, hi, 0, 1, 2, 3, 4, 5, 6, 7,
                                            8, 9, 10, 11, 12, 13, 14, 15);
      acc = __builtin_amdgcn_wmma_f32_16x16x32_bf16(false, afrag[k], false, bfrag,
                                                    (short)0, acc, false, false);
    }
    // C/D layout: lane<16 -> N=lane, M=r; lane>=16 -> N=lane-16, M=8+r
    unsigned short* drow = D + (size_t)(rb * 16 + half * 8) * FOUT + nt * 16 + m;
#pragma unroll
    for (int r = 0; r < 8; ++r) drow[(size_t)r * FOUT] = f2bf(acc[r]);
  }
}

// ---------------------------- edge scatter ---------------------------------
// agg[dst] += h[src] * dinv[src]*dinv[dst];  4 features per thread.
template <int F>
__global__ __launch_bounds__(256) void scatter_edges(
    const long long* __restrict__ src, const long long* __restrict__ dst,
    const float* __restrict__ dinv, const unsigned short* __restrict__ hb,
    float* __restrict__ agg, long long E) {
  constexpr int TPE = F / 4;
  long long g = (long long)blockIdx.x * blockDim.x + threadIdx.x;
  long long e = g / TPE;
  if (e >= E) return;
  int fo = (int)(g - e * TPE) * 4;
  if (e + 4096 < E) {                         // stream the edge list ahead
    __builtin_prefetch(src + e + 4096, 0, 1); // -> global_prefetch_b8
    __builtin_prefetch(dst + e + 4096, 0, 1);
  }
  int s = (int)src[e];
  int d = (int)dst[e];
  float norm = dinv[s] * dinv[d];
  uint2 pk = *(const uint2*)(hb + (size_t)s * F + fo);
  float* ap = agg + (size_t)d * F + fo;
  atomicAdd(ap + 0, bf2f((unsigned short)(pk.x & 0xffff)) * norm);
  atomicAdd(ap + 1, bf2f((unsigned short)(pk.x >> 16)) * norm);
  atomicAdd(ap + 2, bf2f((unsigned short)(pk.y & 0xffff)) * norm);
  atomicAdd(ap + 3, bf2f((unsigned short)(pk.y >> 16)) * norm);
}

// ---------------------------- layer epilogue -------------------------------
// h = relu(agg + t*dinv^2 + b [+ s + bsk]);  emit bf16 and/or f32.
template <int F>
__global__ void finish_layer(const float* agg, const unsigned short* __restrict__ t,
                             const unsigned short* __restrict__ s,
                             const float* __restrict__ bias,
                             const float* __restrict__ bias_sk,
                             const float* __restrict__ dinv,
                             unsigned short* __restrict__ out_bf, float* out_f, int n) {
  int g = blockIdx.x * blockDim.x + threadIdx.x;
  if (g >= n * F) return;
  int i = g / F;
  int f = g - i * F;
  float di = dinv[i];
  float v = agg[g] + bf2f(t[g]) * (di * di) + bias[f];
  if (s) v += bf2f(s[g]) + bias_sk[f];
  v = fmaxf(v, 0.0f);
  if (out_bf) out_bf[g] = f2bf(v);
  if (out_f)  out_f[g]  = v;
}

// ---------------------------- head -----------------------------------------
__global__ void final_head(const float* __restrict__ h3, const float* __restrict__ wlin,
                           const float* __restrict__ blin, float* __restrict__ out, int n) {
  int i = blockIdx.x * blockDim.x + threadIdx.x;
  if (i >= n) return;
  const float* hp = h3 + (size_t)i * 16;
  float acc = blin[0];
#pragma unroll
  for (int f = 0; f < 16; ++f) acc += hp[f] * wlin[f];
  out[i] = 1.0f / (1.0f + __expf(-acc));
}

// ---------------------------------------------------------------------------
extern "C" void kernel_launch(void* const* d_in, const int* in_sizes, int n_in,
                              void* d_out, int out_size, void* d_ws, size_t ws_size,
                              hipStream_t stream) {
  (void)n_in; (void)out_size; (void)ws_size;
  const float*     x     = (const float*)d_in[0];
  const long long* ei    = (const long long*)d_in[1];   // int64 [2, E]
  const float*     W1    = (const float*)d_in[2];
  const float*     b1    = (const float*)d_in[3];
  const float*     W2    = (const float*)d_in[4];
  const float*     b2    = (const float*)d_in[5];
  const float*     W3    = (const float*)d_in[6];
  const float*     b3    = (const float*)d_in[7];
  const float*     Wsk02 = (const float*)d_in[8];
  const float*     bsk02 = (const float*)d_in[9];
  const float*     Wsk13 = (const float*)d_in[10];
  const float*     bsk13 = (const float*)d_in[11];
  const float*     Wlin  = (const float*)d_in[12];
  const float*     blin  = (const float*)d_in[13];

  const long long N = in_sizes[0] / 128;   // 200000
  const long long E = in_sizes[1] / 2;     // 6400000
  const long long* src = ei;
  const long long* dst = ei + E;

  char*  ws  = (char*)d_ws;
  size_t off = 0;
  auto take = [&](size_t bytes) -> char* {
    char* p = ws + off;
    off += (bytes + 255) & ~(size_t)255;
    return p;
  };
  float*          dinv = (float*)take(N * 4);
  unsigned short* xb   = (unsigned short*)take(N * 128 * 2);  // x bf16
  unsigned short* h1b  = (unsigned short*)take(N * 64 * 2);
  unsigned short* h2b  = (unsigned short*)take(N * 32 * 2);
  float*          agg  = (float*)take(N * 64 * 4);            // reused per layer
  unsigned short* tb   = (unsigned short*)take(N * 64 * 2);   // x@W (reused)
  unsigned short* sb   = (unsigned short*)take(N * 32 * 2);   // skip GEMM (reused)
  unsigned short* W1t    = (unsigned short*)take(128 * 64 * 2);
  unsigned short* W2t    = (unsigned short*)take(64 * 32 * 2);
  unsigned short* W3t    = (unsigned short*)take(32 * 16 * 2);
  unsigned short* Wsk02t = (unsigned short*)take(128 * 32 * 2);
  unsigned short* Wsk13t = (unsigned short*)take(64 * 16 * 2);

  const int BT = 256;
  auto blocks = [&](long long work) { return (unsigned)((work + BT - 1) / BT); };

  // degrees -> dinv (in place)
  deg_init<<<blocks(N), BT, 0, stream>>>(dinv, (int)N);
  deg_accum<<<blocks(E), BT, 0, stream>>>(dst, dinv, E);
  deg_to_dinv<<<blocks(N), BT, 0, stream>>>(dinv, (int)N);

  // precision conversions
  cvt_f32_to_bf16<<<blocks((N * 128 + 1) / 2), BT, 0, stream>>>(x, xb, N * 128);
  transpose_w<<<blocks(128 * 64), BT, 0, stream>>>(W1, W1t, 128, 64);
  transpose_w<<<blocks(64 * 32), BT, 0, stream>>>(W2, W2t, 64, 32);
  transpose_w<<<blocks(32 * 16), BT, 0, stream>>>(W3, W3t, 32, 16);
  transpose_w<<<blocks(128 * 32), BT, 0, stream>>>(Wsk02, Wsk02t, 128, 32);
  transpose_w<<<blocks(64 * 16), BT, 0, stream>>>(Wsk13, Wsk13t, 64, 16);

  const unsigned gemmGrid = (unsigned)(((N + 15) / 16 + 3) / 4);  // 4 waves/block

  // -------- layer 1: h1 = relu(GCN(x, W1, b1)) --------
  gemm_bf16_wmma<128, 64><<<gemmGrid, 128, 0, stream>>>(xb, W1t, tb, (int)N);
  zero_f32<<<blocks(N * 64), BT, 0, stream>>>(agg, N * 64);
  scatter_edges<64><<<blocks(E * 16), BT, 0, stream>>>(src, dst, dinv, tb, agg, E);
  finish_layer<64><<<blocks(N * 64), BT, 0, stream>>>(agg, tb, nullptr, b1, nullptr,
                                                      dinv, h1b, nullptr, (int)N);

  // -------- layer 2: h2 = relu(GCN(h1, W2, b2) + x@Wsk02 + bsk02) --------
  gemm_bf16_wmma<64, 32><<<gemmGrid, 128, 0, stream>>>(h1b, W2t, tb, (int)N);
  gemm_bf16_wmma<128, 32><<<gemmGrid, 128, 0, stream>>>(xb, Wsk02t, sb, (int)N);
  zero_f32<<<blocks(N * 32), BT, 0, stream>>>(agg, N * 32);
  scatter_edges<32><<<blocks(E * 8), BT, 0, stream>>>(src, dst, dinv, tb, agg, E);
  finish_layer<32><<<blocks(N * 32), BT, 0, stream>>>(agg, tb, sb, b2, bsk02,
                                                      dinv, h2b, nullptr, (int)N);

  // -------- layer 3: h3 = relu(GCN(h2, W3, b3) + h1@Wsk13 + bsk13) --------
  gemm_bf16_wmma<32, 16><<<gemmGrid, 128, 0, stream>>>(h2b, W3t, tb, (int)N);
  gemm_bf16_wmma<64, 16><<<gemmGrid, 128, 0, stream>>>(h1b, Wsk13t, sb, (int)N);
  zero_f32<<<blocks(N * 16), BT, 0, stream>>>(agg, N * 16);
  scatter_edges<16><<<blocks(E * 4), BT, 0, stream>>>(src, dst, dinv, tb, agg, E);
  finish_layer<16><<<blocks(N * 16), BT, 0, stream>>>(agg, tb, sb, b3, bsk13,
                                                      dinv, nullptr, agg, (int)N);

  // -------- head: sigmoid(h3 @ Wlin + blin) --------
  final_head<<<blocks(N), BT, 0, stream>>>(agg, Wlin, blin, (float*)d_out, (int)N);
}
